// MyOutRGCN_687194767721
// MI455X (gfx1250) — matrix operand order repaired and checked by reading, not compiled, not run on hardware
//
#include <hip/hip_runtime.h>
#include <math.h>

// ---------------------------------------------------------------------------
// RGCN (2 relations, mean aggr) restructured as:
//   scatter-add features per relation  ->  one fused K=384 WMMA GEMM per layer
// Layer:  out = leakyrelu( bias + [h | S0*inv0 | S1*inv1] @ [root; W0; W1] )
// ---------------------------------------------------------------------------

typedef __attribute__((ext_vector_type(16))) _Float16 v16h;
typedef __attribute__((ext_vector_type(8)))  _Float16 v8h;
typedef __attribute__((ext_vector_type(8)))  float    v8f;

#define C_IN  128
#define NREL  2
#define KTOT  (C_IN * (NREL + 1))   // 384

// ---------------- utility: zero fill (graph-capture safe) -------------------
__global__ void zero_f32(float* __restrict__ p, long n) {
    long i = (long)blockIdx.x * blockDim.x + threadIdx.x;
    if (i < n) p[i] = 0.0f;
}

// ---------------- per-(relation,dst) in-degree, computed once ---------------
__global__ void count_edges(const int* __restrict__ ei, const int* __restrict__ et,
                            float* __restrict__ cnt, int E, int N) {
    int e = blockIdx.x * blockDim.x + threadIdx.x;
    if (e >= E) return;
    int dst = ei[E + e];
    int r   = et[e];
    atomicAdd(&cnt[r * N + dst], 1.0f);
}

__global__ void invert_cnt(float* __restrict__ cnt, int n) {
    int i = blockIdx.x * blockDim.x + threadIdx.x;
    if (i < n) cnt[i] = 1.0f / fmaxf(cnt[i], 1.0f);
}

// -------- edge scatter: S[r][dst] += h[src]  (1 wave lane = 4 channels) -----
__global__ void scatter_edges(const float* __restrict__ h, const int* __restrict__ ei,
                              const int* __restrict__ et, float* __restrict__ S,
                              int E, int N) {
    int t = blockIdx.x * blockDim.x + threadIdx.x;
    int e = t >> 5;                 // 32 lanes per edge
    if (e >= E) return;
    int lane = t & 31;
    int src = ei[e];
    int dst = ei[E + e];
    int r   = et[e];
    const float4 v = *(const float4*)(h + (long)src * C_IN + lane * 4);
    float* p = S + ((long)r * N + dst) * C_IN + lane * 4;
    atomicAdd(p + 0, v.x);
    atomicAdd(p + 1, v.y);
    atomicAdd(p + 2, v.z);
    atomicAdd(p + 3, v.w);
}

// -------- build augmented f16 A = [h | S0*inv0 | S1*inv1], row stride 384 ---
__global__ void finalize_a(const float* __restrict__ h, const float* __restrict__ S,
                           const float* __restrict__ inv, _Float16* __restrict__ A,
                           int N) {
    long i = (long)blockIdx.x * blockDim.x + threadIdx.x;
    if (i >= (long)N * KTOT) return;
    int v = (int)(i / KTOT);
    int k = (int)(i % KTOT);
    float val;
    if (k < C_IN) {
        val = h[(long)v * C_IN + k];
    } else {
        int r = (k - C_IN) / C_IN;          // 0 or 1
        int c = k & (C_IN - 1);
        val = S[((long)r * N + v) * C_IN + c] * inv[r * N + v];
    }
    A[i] = (_Float16)val;
}

// -------- build transposed f16 B: Bt[n][k] = [root;W0;W1][k][n] -------------
__global__ void build_bt(const float* __restrict__ root, const float* __restrict__ W,
                         _Float16* __restrict__ Bt) {
    int i = blockIdx.x * blockDim.x + threadIdx.x;
    if (i >= C_IN * KTOT) return;
    int n = i / KTOT;
    int k = i % KTOT;
    float v = (k < C_IN) ? root[k * C_IN + n]
                         : W[(long)(k - C_IN) * C_IN + n];  // W is [2][128][128] contiguous
    Bt[i] = (_Float16)v;
}

// -------- fused WMMA GEMM: out[N,128] = lrelu(A[N,384] @ B[384,128] + bias) -
// 1 wave -> one 16x16 tile; 8 waves/block cover all 128 output columns.
__global__ void rgcn_gemm(const _Float16* __restrict__ A, const _Float16* __restrict__ Bt,
                          const float* __restrict__ bias, float* __restrict__ out, int N) {
    const int lane = threadIdx.x & 31;
    const int wave = threadIdx.x >> 5;          // 0..7 : output column tile
    const int hi   = lane >> 4;                 // lane group (0 | 1)
    const int m0   = blockIdx.x * 16;
    int row = m0 + (lane & 15);
    if (row >= N) row = N - 1;                  // clamp, no divergence (EXEC all-1 for WMMA)
    const int n = wave * 16 + (lane & 15);

    // A fragment per ISA layout: lanes<16 hold K {0..7}U{16..23}, lanes>=16 +8
    const _Float16* arow = A  + (long)row * KTOT + hi * 8;
    // B fragment (same K order as A): lanes<16 hold K {0..7}U{16..23}, lanes>=16 +8; from Bt[n][k]
    const _Float16* brow = Bt + (long)n   * KTOT + hi * 8;

    v8f acc = {};
    union { v16h v; v8h h[2]; } ua, ub;
#pragma unroll
    for (int kb = 0; kb < KTOT / 32; ++kb) {
        ua.h[0] = *(const v8h*)(arow + kb * 32);
        ua.h[1] = *(const v8h*)(arow + kb * 32 + 16);
        ub.h[0] = *(const v8h*)(brow + kb * 32);
        ub.h[1] = *(const v8h*)(brow + kb * 32 + 16);
        acc = __builtin_amdgcn_wmma_f32_16x16x32_f16(
            /*neg_a=*/false, ua.v, /*neg_b=*/false, ub.v,
            /*c_mod=*/(short)0, acc, /*reuse_a=*/false, /*reuse_b=*/false);
    }

    const float bn = bias[n];
#pragma unroll
    for (int j = 0; j < 8; ++j) {               // C/D layout: VGPR j -> M = j + 8*hi
        int m = m0 + j + hi * 8;
        if (m < N) {
            float v = acc[j] + bn;
            out[(long)m * C_IN + n] = fmaxf(v, 0.0f) + 0.01f * fminf(v, 0.0f);
        }
    }
}

// -------- h_sel = h2[idx]  ---------------------------------------------------
__global__ void gather_rows(const float* __restrict__ h, const int* __restrict__ idx,
                            float* __restrict__ out, int SEL) {
    int i = blockIdx.x * blockDim.x + threadIdx.x;
    if (i >= SEL * C_IN) return;
    int s = i / C_IN;
    int c = i & (C_IN - 1);
    out[i] = h[(long)idx[s] * C_IN + c];
}

// -------- tiny MLP head: sigmoid(h_sel @ mlp_w + mlp_b), OUT=5 --------------
__global__ void mlp_sigmoid(const float* __restrict__ hsel, const float* __restrict__ w,
                            const float* __restrict__ b, float* __restrict__ out,
                            int SEL, int OUT) {
    int i = blockIdx.x * blockDim.x + threadIdx.x;
    if (i >= SEL * OUT) return;
    int s = i / OUT;
    int o = i % OUT;
    float acc = b[o];
#pragma unroll 16
    for (int c = 0; c < C_IN; ++c)
        acc += hsel[(long)s * C_IN + c] * w[c * OUT + o];
    out[i] = 1.0f / (1.0f + expf(-acc));
}

// ---------------------------------------------------------------------------
extern "C" void kernel_launch(void* const* d_in, const int* in_sizes, int n_in,
                              void* d_out, int out_size, void* d_ws, size_t ws_size,
                              hipStream_t stream) {
    const float* x     = (const float*)d_in[0];
    const int*   ei    = (const int*)  d_in[1];
    const int*   et    = (const int*)  d_in[2];
    const int*   idx   = (const int*)  d_in[3];
    const float* W1    = (const float*)d_in[4];
    const float* root1 = (const float*)d_in[5];
    const float* b1    = (const float*)d_in[6];
    const float* W2    = (const float*)d_in[7];
    const float* root2 = (const float*)d_in[8];
    const float* b2    = (const float*)d_in[9];
    const float* mlpw  = (const float*)d_in[10];
    const float* mlpb  = (const float*)d_in[11];
    float* out = (float*)d_out;

    const int N   = in_sizes[0] / C_IN;   // 50000
    const int E   = in_sizes[2];          // 1000000
    const int SEL = in_sizes[3];          // 5000
    const int OUT = in_sizes[11];         // 5

    // ---- workspace carve-up (256B aligned regions) ----
    char* ws = (char*)d_ws;
    size_t off = 0;
    auto take = [&](size_t bytes) { char* p = ws + off; off = (off + bytes + 255) & ~(size_t)255; return p; };
    float*    S   = (float*)   take((size_t)NREL * N * C_IN * sizeof(float));   // 51.2 MB
    float*    inv = (float*)   take((size_t)NREL * N * sizeof(float));          // 0.4 MB (cnt -> 1/cnt)
    _Float16* Aug = (_Float16*)take((size_t)N * KTOT * sizeof(_Float16));       // 38.4 MB
    _Float16* Bt  = (_Float16*)take((size_t)C_IN * KTOT * sizeof(_Float16));    // 96 KB
    float*    h1  = (float*)   take((size_t)N * C_IN * sizeof(float));          // 25.6 MB
    float*    h2  = (float*)   take((size_t)N * C_IN * sizeof(float));          // 25.6 MB
    (void)ws_size;

    const int TB = 256;
    const long nS   = (long)NREL * N * C_IN;
    const long nAug = (long)N * KTOT;

    // ---- degree (static across layers: computed once) ----
    zero_f32   <<<(NREL * N + TB - 1) / TB, TB, 0, stream>>>(inv, NREL * N);
    count_edges<<<(E + TB - 1) / TB,        TB, 0, stream>>>(ei, et, inv, E, N);
    invert_cnt <<<(NREL * N + TB - 1) / TB, TB, 0, stream>>>(inv, NREL * N);

    // ---- layer 1 ----
    zero_f32     <<<(int)((nS + TB - 1) / TB), TB, 0, stream>>>(S, nS);
    scatter_edges<<<(int)(((long)E * 32 + TB - 1) / TB), TB, 0, stream>>>(x, ei, et, S, E, N);
    finalize_a   <<<(int)((nAug + TB - 1) / TB), TB, 0, stream>>>(x, S, inv, Aug, N);
    build_bt     <<<(C_IN * KTOT + TB - 1) / TB, TB, 0, stream>>>(root1, W1, Bt);
    rgcn_gemm    <<<(N + 15) / 16, TB, 0, stream>>>(Aug, Bt, b1, h1, N);

    // ---- layer 2 ----
    zero_f32     <<<(int)((nS + TB - 1) / TB), TB, 0, stream>>>(S, nS);
    scatter_edges<<<(int)(((long)E * 32 + TB - 1) / TB), TB, 0, stream>>>(h1, ei, et, S, E, N);
    finalize_a   <<<(int)((nAug + TB - 1) / TB), TB, 0, stream>>>(h1, S, inv, Aug, N);
    build_bt     <<<(C_IN * KTOT + TB - 1) / TB, TB, 0, stream>>>(root2, W2, Bt);
    rgcn_gemm    <<<(N + 15) / 16, TB, 0, stream>>>(Aug, Bt, b2, h2, N);

    // ---- head: gather selected rows, tiny sigmoid MLP ----
    gather_rows<<<(SEL * C_IN + TB - 1) / TB, TB, 0, stream>>>(h2, idx, out, SEL);
    mlp_sigmoid<<<(SEL * OUT + TB - 1) / TB, TB, 0, stream>>>(out, mlpw, mlpb,
                                                              out + (size_t)SEL * C_IN, SEL, OUT);
}
